// NodeGTransformerBlocks_43181601194865
// MI455X (gfx1250) — matrix operand order repaired and checked by reading, not compiled, not run on hardware
//
#include <hip/hip_runtime.h>
#include <hip/hip_bf16.h>
#include <cstdint>

typedef __attribute__((ext_vector_type(16))) _Float16 v16h;
typedef __attribute__((ext_vector_type(8)))  _Float16 v8h;
typedef __attribute__((ext_vector_type(8)))  float    v8f;

#define NTOK  3072
#define UNITS 1024
#define HEADS 16
#define HDIM  64
#define CHUNK 64
#define NEGBIG (-1.0e9f)

// ---------------------------------------------------------------------------
// Fragment loaders (layouts per CDNA5 ISA 7.12.2, wave32)
// A 16x32 f16: lane holds row = lane&15; K in {kb..kb+7} U {kb+16..kb+23},
//              kb = (lane>>4)*8. Two contiguous 16B chunks.
// B 32x16 f16 (from K-contiguous N x K storage): lane holds col = lane&15,
//              K in {kb..kb+7} U {kb+16..kb+23}, kb = (lane>>4)*8.
// C/D 16x16 f32: element i -> row = i + (lane>>4)*8, col = lane&15.
// ---------------------------------------------------------------------------
__device__ __forceinline__ v16h load_a_frag(const _Float16* base, int lda) {
  const int lane = threadIdx.x & 31;
  const int row  = lane & 15;
  const int kb   = (lane >> 4) << 3;
  const _Float16* p = base + row * lda + kb;
  v8h lo = *(const v8h*)(p);
  v8h hi = *(const v8h*)(p + 16);
  v16h r;
#pragma unroll
  for (int i = 0; i < 8; ++i) { r[i] = lo[i]; r[i + 8] = hi[i]; }
  return r;
}

__device__ __forceinline__ v16h load_b_frag(const _Float16* baseT, int ldb) {
  const int lane = threadIdx.x & 31;
  const int col  = lane & 15;
  const int kb   = (lane >> 4) << 3;
  const _Float16* p = baseT + col * ldb + kb;
  v8h lo = *(const v8h*)(p);
  v8h hi = *(const v8h*)(p + 16);
  v16h r;
#pragma unroll
  for (int i = 0; i < 8; ++i) { r[i] = lo[i]; r[i + 8] = hi[i]; }
  return r;
}

__device__ __forceinline__ v8f wmma_f16(v16h a, v16h b, v8f c) {
  return __builtin_amdgcn_wmma_f32_16x16x32_f16(false, a, false, b,
                                                (short)0, c, false, false);
}

// Async global -> LDS copy of one 16B element per call (gfx1250 ASYNCcnt path)
__device__ __forceinline__ void async_cp16(unsigned ldsOff,
                                           const _Float16* gsrc) {
  asm volatile("global_load_async_to_lds_b128 %0, %1, off"
               :: "v"(ldsOff), "v"((unsigned long long)(uintptr_t)gsrc)
               : "memory");
}

// ---------------------------------------------------------------------------
// Kernel 1: convert x f32 -> f16 (row major, 3072x1024)
// ---------------------------------------------------------------------------
__global__ void k_cvt_x(const float* __restrict__ x, _Float16* __restrict__ x16,
                        int n) {
  int i = blockIdx.x * blockDim.x + threadIdx.x;
  if (i < n) x16[i] = (_Float16)x[i];
}

// Kernel 2: W [K x N] f32  ->  WT [N x K] f16 (so B-fragments are contiguous)
__global__ void k_transpose_w(const float* __restrict__ w,
                              _Float16* __restrict__ wt) {
  int i = blockIdx.x * blockDim.x + threadIdx.x;   // UNITS*UNITS threads
  int k = i >> 10;
  int n = i & (UNITS - 1);
  wt[n * UNITS + k] = (_Float16)w[k * UNITS + n];
}

// ---------------------------------------------------------------------------
// Kernel 3: QKV GEMM. One 16x64 tile per wave (one A-frag feeds 4 WMMAs),
// 128 WMMA over K=1024. A 64-wide U tile == exactly one head.
// z=0: Q -> [h][n][d] f16, scaled by 1/sqrt(64)
// z=1: K -> [h][n][d] f16
// z=2: V -> [h][d][n] f16 (transposed so PV B-fragments are contiguous)
// ---------------------------------------------------------------------------
__global__ __launch_bounds__(256) void k_qkv_gemm(
    const _Float16* __restrict__ x16, const _Float16* __restrict__ WqT,
    const _Float16* __restrict__ WkT, const _Float16* __restrict__ WvT,
    _Float16* __restrict__ Qb, _Float16* __restrict__ Kb,
    _Float16* __restrict__ Vt) {
  const int wave  = threadIdx.x >> 5;
  const int tile  = blockIdx.x * (blockDim.x >> 5) + wave;  // 3072 tiles
  const int qTile = tile >> 4;        // / 16 u-supertiles
  const int uSup  = tile & 15;
  const int qBase = qTile * 16;
  const int uBase = uSup * 64;        // head-aligned
  const int z     = blockIdx.z;
  const _Float16* WT    = (z == 0) ? WqT : (z == 1) ? WkT : WvT;
  const _Float16* aBase = x16 + qBase * UNITS;
  const _Float16* bBase = WT + uBase * UNITS;

  v8f c0 = {}, c1 = {}, c2 = {}, c3 = {};
#pragma unroll 2
  for (int kk = 0; kk < UNITS / 32; ++kk) {
    __builtin_prefetch(bBase + (kk + 1) * 32, 0, 3);
    v16h a  = load_a_frag(aBase + kk * 32, UNITS);
    v16h b0 = load_b_frag(bBase + 0 * 16 * UNITS + kk * 32, UNITS);
    v16h b1 = load_b_frag(bBase + 1 * 16 * UNITS + kk * 32, UNITS);
    v16h b2 = load_b_frag(bBase + 2 * 16 * UNITS + kk * 32, UNITS);
    v16h b3 = load_b_frag(bBase + 3 * 16 * UNITS + kk * 32, UNITS);
    c0 = wmma_f16(a, b0, c0);
    c1 = wmma_f16(a, b1, c1);
    c2 = wmma_f16(a, b2, c2);
    c3 = wmma_f16(a, b3, c3);
  }

  const int lane = threadIdx.x & 31;
  const int col  = lane & 15;
  const int rOff = (lane >> 4) << 3;
  const int h    = uBase >> 6;                    // whole tile is one head
  v8f cc[4] = {c0, c1, c2, c3};
#pragma unroll
  for (int j = 0; j < 4; ++j) {
    const int d0 = j * 16 + col;
#pragma unroll
    for (int i = 0; i < 8; ++i) {
      const int nrow = qBase + rOff + i;
      if (z == 0)
        Qb[(h * NTOK + nrow) * HDIM + d0] = (_Float16)(cc[j][i] * 0.125f);
      else if (z == 1)
        Kb[(h * NTOK + nrow) * HDIM + d0] = (_Float16)cc[j][i];
      else
        Vt[(h * HDIM + d0) * NTOK + nrow] = (_Float16)cc[j][i];
    }
  }
}

// ---------------------------------------------------------------------------
// Kernel 4: block-masked flash attention. 4 waves/block share one head;
// K/V 64-key chunks staged once per block into LDS with async copies,
// double-buffered on ASYNCcnt. 8 WMMA scores + 8 WMMA P*V per chunk per wave.
// ---------------------------------------------------------------------------
__global__ __launch_bounds__(128) void k_attention(
    const _Float16* __restrict__ Qb, const _Float16* __restrict__ Kb,
    const _Float16* __restrict__ Vt, const int* __restrict__ block_ids,
    const float* __restrict__ nodal_mask, _Float16* __restrict__ attn) {
  __shared__ __align__(32) _Float16 kTile[2][CHUNK * HDIM];   // [key][d]
  __shared__ __align__(32) _Float16 vTile[2][HDIM * CHUNK];   // [d][key]
  __shared__ __align__(32) _Float16 pShared[4][16 * CHUNK];

  const int tid   = threadIdx.x;
  const int wave  = tid >> 5;
  const int lane  = tid & 31;
  const int h     = blockIdx.x;
  const int qTile = blockIdx.y * 4 + wave;
  const int qBase = qTile * 16;

  const _Float16* Qh = Qb + h * NTOK * HDIM + qBase * HDIM;
  const _Float16* Kh = Kb + h * NTOK * HDIM;
  const _Float16* Vh = Vt + h * HDIM * NTOK;
  _Float16* pLds = pShared[wave];

  // Stage one 64-key K/V chunk into LDS buffer `buf` (8 async ops / thread).
  auto stage = [&](int kc, int buf) {
    const int kcBase = kc * CHUNK;
    // K chunk: contiguous 8KB ([key][d], ld=HDIM)
#pragma unroll
    for (int p = 0; p < 4; ++p) {
      const int e = p * 1024 + tid * 8;              // half index
      async_cp16((unsigned)(uintptr_t)&kTile[buf][e], Kh + kcBase * HDIM + e);
    }
    // V chunk: 64 rows of 128B ([d][key], global ld=NTOK -> LDS ld=CHUNK)
#pragma unroll
    for (int p = 0; p < 4; ++p) {
      const int c   = p * 128 + tid;                 // 16B chunk id (0..511)
      const int d   = c >> 3;
      const int off = (c & 7) * 8;
      async_cp16((unsigned)(uintptr_t)&vTile[buf][d * CHUNK + off],
                 Vh + d * NTOK + kcBase + off);
    }
  };

  stage(0, 0);

  const v16h aq0 = load_a_frag(Qh, HDIM);        // d = 0..31
  const v16h aq1 = load_a_frag(Qh + 32, HDIM);   // d = 32..63

  const int col  = lane & 15;
  const int rOff = (lane >> 4) << 3;
  const int bidQ = block_ids[qBase + col];       // lane L holds row L&15's id

  float m[8], l[8];
#pragma unroll
  for (int i = 0; i < 8; ++i) { m[i] = -1.0e30f; l[i] = 0.0f; }
  v8f o0 = {}, o1 = {}, o2 = {}, o3 = {};

  for (int kc = 0; kc < NTOK / CHUNK; ++kc) {
    const int cur = kc & 1;
    if (kc + 1 < NTOK / CHUNK) {
      stage(kc + 1, cur ^ 1);                        // prefetch next chunk
      asm volatile("s_wait_asynccnt 8" ::: "memory");  // current 8 retired
    } else {
      asm volatile("s_wait_asynccnt 0" ::: "memory");
    }
    __syncthreads();                                 // publish LDS tiles
    const _Float16* kT = kTile[cur];
    const _Float16* vT = vTile[cur];
    const int kcBase = kc * CHUNK;

    v8f st[4];
#pragma unroll
    for (int t = 0; t < 4; ++t) {
      v16h bk0 = load_b_frag(kT + t * 16 * HDIM, HDIM);       // d 0..31
      v16h bk1 = load_b_frag(kT + t * 16 * HDIM + 32, HDIM);  // d 32..63
      v8f s = {};
      s = wmma_f16(aq0, bk0, s);
      s = wmma_f16(aq1, bk1, s);
      const int   kg      = kcBase + t * 16 + col;
      const int   bidK    = block_ids[kg];
      const float maskAdd = (1.0f - nodal_mask[kg]) * NEGBIG;
#pragma unroll
      for (int i = 0; i < 8; ++i) {
        const int blo = __shfl(bidQ, i, 32);       // block id of row i
        const int bhi = __shfl(bidQ, i + 8, 32);   // block id of row i+8
        const int bq  = (lane < 16) ? blo : bhi;
        s[i] += ((bq == bidK) ? 0.0f : NEGBIG) + maskAdd;
      }
      st[t] = s;
    }

    // ---- online softmax over this 64-key chunk (row reductions via shfl) --
#pragma unroll
    for (int i = 0; i < 8; ++i) {
      float lm = fmaxf(fmaxf(st[0][i], st[1][i]), fmaxf(st[2][i], st[3][i]));
      lm = fmaxf(lm, __shfl_xor(lm, 8, 32));
      lm = fmaxf(lm, __shfl_xor(lm, 4, 32));
      lm = fmaxf(lm, __shfl_xor(lm, 2, 32));
      lm = fmaxf(lm, __shfl_xor(lm, 1, 32));
      const float mNew = fmaxf(m[i], lm);
      float p[4], rs = 0.0f;
#pragma unroll
      for (int t = 0; t < 4; ++t) { p[t] = __expf(st[t][i] - mNew); rs += p[t]; }
      rs += __shfl_xor(rs, 8, 32);
      rs += __shfl_xor(rs, 4, 32);
      rs += __shfl_xor(rs, 2, 32);
      rs += __shfl_xor(rs, 1, 32);
      const float alpha = __expf(m[i] - mNew);
      l[i] = l[i] * alpha + rs;
      m[i] = mNew;
      o0[i] *= alpha; o1[i] *= alpha; o2[i] *= alpha; o3[i] *= alpha;
      // D-layout -> row-major P in LDS (16 x 64, f16)
      const int prow = rOff + i;
#pragma unroll
      for (int t = 0; t < 4; ++t)
        pLds[prow * CHUNK + t * 16 + col] = (_Float16)p[t];
    }
    asm volatile("s_wait_dscnt 0" ::: "memory");

    // ---- P (16x64) * V (64x64): P as A-fragments, V from LDS -------------
    v16h ap0 = load_a_frag(pLds, CHUNK);         // keys 0..31 of chunk
    v16h ap1 = load_a_frag(pLds + 32, CHUNK);    // keys 32..63 of chunk
#pragma unroll
    for (int dt = 0; dt < 4; ++dt) {
      v16h bvA = load_b_frag(vT + dt * 16 * CHUNK, CHUNK);
      v16h bvB = load_b_frag(vT + dt * 16 * CHUNK + 32, CHUNK);
      v8f* o = (dt == 0) ? &o0 : (dt == 1) ? &o1 : (dt == 2) ? &o2 : &o3;
      *o = wmma_f16(ap0, bvA, *o);
      *o = wmma_f16(ap1, bvB, *o);
    }
    __syncthreads();   // all waves done reading before buffer is re-staged
  }

  // normalize and write attn [n][h*64+d] f16 (input to final GEMM)
#pragma unroll
  for (int i = 0; i < 8; ++i) {
    const int q = qBase + rOff + i;
    const float inv = 1.0f / l[i];
    _Float16* row = attn + q * UNITS + h * HDIM + col;
    row[0]  = (_Float16)(o0[i] * inv);
    row[16] = (_Float16)(o1[i] * inv);
    row[32] = (_Float16)(o2[i] * inv);
    row[48] = (_Float16)(o3[i] * inv);
  }
}

// ---------------------------------------------------------------------------
// Kernel 5: out = attn(f16) @ Wo -> f32. 16x64 tile per wave.
// ---------------------------------------------------------------------------
__global__ __launch_bounds__(256) void k_out_gemm(
    const _Float16* __restrict__ attn, const _Float16* __restrict__ WoT,
    float* __restrict__ out) {
  const int wave  = threadIdx.x >> 5;
  const int tile  = blockIdx.x * (blockDim.x >> 5) + wave;  // 3072 tiles
  const int qTile = tile >> 4;
  const int uSup  = tile & 15;
  const int qBase = qTile * 16;
  const int uBase = uSup * 64;
  const _Float16* aBase = attn + qBase * UNITS;
  const _Float16* bBase = WoT + uBase * UNITS;

  v8f c0 = {}, c1 = {}, c2 = {}, c3 = {};
#pragma unroll 2
  for (int kk = 0; kk < UNITS / 32; ++kk) {
    __builtin_prefetch(bBase + (kk + 1) * 32, 0, 3);
    v16h a  = load_a_frag(aBase + kk * 32, UNITS);
    v16h b0 = load_b_frag(bBase + 0 * 16 * UNITS + kk * 32, UNITS);
    v16h b1 = load_b_frag(bBase + 1 * 16 * UNITS + kk * 32, UNITS);
    v16h b2 = load_b_frag(bBase + 2 * 16 * UNITS + kk * 32, UNITS);
    v16h b3 = load_b_frag(bBase + 3 * 16 * UNITS + kk * 32, UNITS);
    c0 = wmma_f16(a, b0, c0);
    c1 = wmma_f16(a, b1, c1);
    c2 = wmma_f16(a, b2, c2);
    c3 = wmma_f16(a, b3, c3);
  }

  const int lane = threadIdx.x & 31;
  const int col  = lane & 15;
  const int rOff = (lane >> 4) << 3;
  v8f cc[4] = {c0, c1, c2, c3};
#pragma unroll
  for (int j = 0; j < 4; ++j)
#pragma unroll
    for (int i = 0; i < 8; ++i)
      out[(qBase + rOff + i) * UNITS + uBase + j * 16 + col] = cc[j][i];
}

// ---------------------------------------------------------------------------
extern "C" void kernel_launch(void* const* d_in, const int* in_sizes, int n_in,
                              void* d_out, int out_size, void* d_ws,
                              size_t ws_size, hipStream_t stream) {
  (void)in_sizes; (void)n_in; (void)out_size; (void)ws_size;
  const float* x          = (const float*)d_in[0];
  const float* Wq         = (const float*)d_in[1];
  const float* Wk         = (const float*)d_in[2];
  const float* Wv         = (const float*)d_in[3];
  const float* Wo         = (const float*)d_in[4];
  const float* nodal_mask = (const float*)d_in[5];
  const int*   block_ids  = (const int*)d_in[6];
  float* out = (float*)d_out;

  // workspace layout (f16): ~40 MB total
  _Float16* ws   = (_Float16*)d_ws;
  _Float16* x16  = ws;                               // 3072*1024
  _Float16* WqT  = x16 + NTOK * UNITS;               // 1024*1024
  _Float16* WkT  = WqT + UNITS * UNITS;
  _Float16* WvT  = WkT + UNITS * UNITS;
  _Float16* WoT  = WvT + UNITS * UNITS;
  _Float16* Qb   = WoT + UNITS * UNITS;              // 16*3072*64
  _Float16* Kb   = Qb + NTOK * UNITS;
  _Float16* Vt   = Kb + NTOK * UNITS;
  _Float16* attn = Vt + NTOK * UNITS;

  const int nX = NTOK * UNITS;
  k_cvt_x<<<(nX + 255) / 256, 256, 0, stream>>>(x, x16, nX);

  const int nW = UNITS * UNITS;
  k_transpose_w<<<nW / 256, 256, 0, stream>>>(Wq, WqT);
  k_transpose_w<<<nW / 256, 256, 0, stream>>>(Wk, WkT);
  k_transpose_w<<<nW / 256, 256, 0, stream>>>(Wv, WvT);
  k_transpose_w<<<nW / 256, 256, 0, stream>>>(Wo, WoT);

  // 192 qTiles * 16 u-supertiles = 3072 wave-tiles, 8 waves/block
  k_qkv_gemm<<<dim3(384, 1, 3), 256, 0, stream>>>(x16, WqT, WkT, WvT, Qb, Kb,
                                                  Vt);

  // 16 heads x 48 blocks x 4 waves -> 192 query tiles per head
  k_attention<<<dim3(HEADS, NTOK / 64), 128, 0, stream>>>(
      Qb, Kb, Vt, block_ids, nodal_mask, attn);

  k_out_gemm<<<384, 256, 0, stream>>>(attn, WoT, out);
}
